// WidthCharTransformerMLP_63754494542654
// MI455X (gfx1250) — matrix operand order repaired and checked by reading, not compiled, not run on hardware
//
#include <hip/hip_runtime.h>

// ---------------------------------------------------------------------------
// MI455X (gfx1250) fused char-transformer encoder + MLP.
// - bf16 WMMA (v_wmma_f32_16x16x32_bf16) for all big GEMMs
// - pooling (linear) commuted before Wo: Wo/W1/W2 run on 2048 rows, not 196608
// - attention (~3% of FLOPs) on VALU with __expf
// - embedding gather staged via GLOBAL_LOAD_ASYNC_TO_LDS_B128 (ASYNCcnt)
// - weight panels prefetched with global_prefetch_b8
// ---------------------------------------------------------------------------

typedef __attribute__((ext_vector_type(16))) __bf16 v16bf;
typedef __attribute__((ext_vector_type(8)))  __bf16 v8bf;
typedef __attribute__((ext_vector_type(8)))  float  v8f;

#define N_WORDSC 8192
#define LCH      24
#define VOCABC   128
#define DMODEL   512
#define NHEAD    8
#define DKH      64
#define N_NAMESC 2048
#define WPN      4
#define DOUTC    256

#define XS 520   // bf16 LDS stride for A-side (K=512) tiles; 1040B rotates banks
#define QS 528   // bf16 LDS stride for q/k/v (N=512) tiles; 1056B rotates banks

// --- A fragment for V_WMMA_F32_16X16X32_BF16 (16x32, 16-bit) ---------------
// lanes 0-15 : row = lane,    K = [kb..kb+16) in v[0:7]
// lanes 16-31: row = lane-16, K = [kb+16..kb+32)
// caller passes p = rowbase + kb + ((lane<16)?0:16); hi chunk is p+8.
__device__ inline v16bf load_afrag(const __bf16* p) {
  v8bf lo = *(const v8bf*)(p);
  v8bf hi = *(const v8bf*)(p + 8);
  v16bf a;
#pragma unroll
  for (int i = 0; i < 8; ++i) { a[i] = lo[i]; a[i + 8] = hi[i]; }
  return a;
}

// --- one wave computes D[32 x 64] = A[32 x 512] * W^T slice ----------------
// Alds: LDS bf16, stride XS. WT: global bf16 [512][512], row n = column n of W
// (so each B-fragment lane reads 16 contiguous K values). Dlds: LDS bf16,
// stride dstride. C layout: VGPR r -> row r (lanes 0-15) / row r+8 (16-31).
__device__ inline void gemm_tile_512(const __bf16* Alds,
                                     const __bf16* __restrict__ WT,
                                     __bf16* Dlds, int dstride,
                                     int wave, int lane, bool act)
{
  const int lm   = lane & 15;
  const int koff = (lane < 16) ? 0 : 16;  // A-side K offset
  const int bko  = (lane < 16) ? 0 : 16;  // B-side K offset
  const int hi8  = (lane >> 4) * 8;
#pragma unroll
  for (int nt = 0; nt < 4; ++nt) {
    const int ncol = wave * 64 + nt * 16 + lm;
    const __bf16* wrow = WT + (size_t)ncol * DMODEL + bko;
    // pull next N-tile's weight row toward the WGP while this K-loop runs
    __builtin_prefetch(wrow + 16 * DMODEL, 0, 1);
    v8f acc0 = {};
    v8f acc1 = {};
    for (int kk = 0; kk < 16; ++kk) {
      const int kb0 = kk * 32;
      v16bf b  = *(const v16bf*)(wrow + kb0);
      v16bf a0 = load_afrag(Alds + lm * XS + kb0 + koff);
      v16bf a1 = load_afrag(Alds + (lm + 16) * XS + kb0 + koff);
      acc0 = __builtin_amdgcn_wmma_f32_16x16x32_bf16(false, a0, false, b,
                                                     (short)0, acc0, false, false);
      acc1 = __builtin_amdgcn_wmma_f32_16x16x32_bf16(false, a1, false, b,
                                                     (short)0, acc1, false, false);
    }
#pragma unroll
    for (int r = 0; r < 8; ++r) {
      float x0 = acc0[r], x1 = acc1[r];
      if (act) { x0 = tanhf(x0); x1 = tanhf(x1); }
      Dlds[(r + hi8) * dstride + ncol]      = (__bf16)x0;
      Dlds[(16 + r + hi8) * dstride + ncol] = (__bf16)x1;
    }
  }
}

// ---------------------------------------------------------------------------
// prep: fp32 -> bf16 converts (weights stored transposed [N][K])
// ---------------------------------------------------------------------------
__global__ void convert_kernel(const float* __restrict__ src,
                               __bf16* __restrict__ dst, int n) {
  int i = blockIdx.x * 256 + threadIdx.x;
  if (i < n) dst[i] = (__bf16)src[i];
}

__global__ void transpose_convert_kernel(const float* __restrict__ src,
                                         __bf16* __restrict__ dst,
                                         int K, int N) {
  // src: [K][N] row-major, dst: [N][K]
  int i = blockIdx.x * 256 + threadIdx.x;
  if (i < K * N) {
    int n = i / K, k = i - n * K;
    dst[(size_t)n * K + k] = (__bf16)src[(size_t)k * N + n];
  }
}

// ---------------------------------------------------------------------------
// encoder: one block per word. embed (async->LDS) -> QKV (WMMA) ->
// VALU attention -> masked mean over chars -> pooled[word][512] fp32
// ---------------------------------------------------------------------------
__global__ __launch_bounds__(256) void encoder_kernel(
    const int*    __restrict__ inputs,   // [8192][24]
    const __bf16* __restrict__ embB,     // [128][512]
    const __bf16* __restrict__ WqT,      // [512][512] (transposed)
    const __bf16* __restrict__ WkT,
    const __bf16* __restrict__ WvT,
    float*        __restrict__ pooled)   // [8192][512]
{
  __shared__ alignas(32) __bf16 xs[32 * XS];
  __shared__ alignas(32) __bf16 qb[32 * QS];
  __shared__ alignas(32) __bf16 kb[32 * QS];
  __shared__ alignas(32) __bf16 vb[32 * QS];
  __shared__ int chars[32];

  const int word = blockIdx.x;
  const int tid  = threadIdx.x;
  const int wave = tid >> 5, lane = tid & 31;

  if (tid < 32) chars[tid] = (tid < LCH) ? inputs[word * LCH + tid] : 0;
  __syncthreads();

  // ---- embed rows 0..23 into LDS via async global->LDS (16B per lane) ----
  // 24 rows x 64 16B-chunks = 1536 chunks; 256 threads -> 6 uniform iterations
  for (int ch = tid; ch < LCH * (DMODEL / 8); ch += 256) {
    const int r = ch >> 6;            // chunk row
    const int c = (ch & 63) * 8;      // chunk col (bf16 elems)
    const unsigned lds_off =
        (unsigned)(unsigned long long)(const void*)(xs + r * XS + c);
    const void* g = (const void*)(embB + (size_t)chars[r] * DMODEL + c);
    asm volatile("global_load_async_to_lds_b128 %0, %1, off"
                 :: "v"(lds_off), "v"(g) : "memory");
  }
  // zero-pad rows 24..31 with plain LDS stores
  for (int i = tid; i < 8 * DMODEL; i += 256) {
    int r = LCH + (i >> 9), c = i & 511;
    xs[r * XS + c] = (__bf16)0.f;
  }
  asm volatile("s_wait_asynccnt 0x0" ::: "memory");
  __syncthreads();

  // ---- Q/K/V projections: 8 waves x 64 cols each, WMMA bf16 ----
  gemm_tile_512(xs, WqT, qb, QS, wave, lane, false);
  gemm_tile_512(xs, WkT, kb, QS, wave, lane, false);
  gemm_tile_512(xs, WvT, vb, QS, wave, lane, false);
  __syncthreads();

  // ---- attention on VALU: one thread per (head, query row) ----
  float sc[LCH];
  const int h  = tid / LCH;
  const int qr = tid - h * LCH;
  const int cbase = h * DKH;
  if (tid < NHEAD * LCH) {
    float mx = -3.0e38f;
#pragma unroll
    for (int j = 0; j < LCH; ++j) {
      float s = 0.f;
      for (int d0 = 0; d0 < DKH; d0 += 8) {
        v8bf qv = *(const v8bf*)(qb + qr * QS + cbase + d0);
        v8bf kv = *(const v8bf*)(kb + j  * QS + cbase + d0);
#pragma unroll
        for (int i = 0; i < 8; ++i) s += (float)qv[i] * (float)kv[i];
      }
      s *= 0.125f;                       // 1/sqrt(64)
      if (chars[j] == 0) s = -1.0e9f;    // key padding mask
      sc[j] = s;
      mx = fmaxf(mx, s);
    }
    float den = 0.f;
#pragma unroll
    for (int j = 0; j < LCH; ++j) { sc[j] = __expf(sc[j] - mx); den += sc[j]; }
    const float inv = 1.f / den;
#pragma unroll
    for (int j = 0; j < LCH; ++j) sc[j] *= inv;
  }
  __syncthreads();   // all q/k reads done; qb becomes the o-buffer

  if (tid < NHEAD * LCH) {
    for (int d0 = 0; d0 < DKH; d0 += 8) {
      float o[8] = {0.f, 0.f, 0.f, 0.f, 0.f, 0.f, 0.f, 0.f};
#pragma unroll
      for (int j = 0; j < LCH; ++j) {
        v8bf vv = *(const v8bf*)(vb + j * QS + cbase + d0);
        const float a = sc[j];
#pragma unroll
        for (int i = 0; i < 8; ++i) o[i] += a * (float)vv[i];
      }
      v8bf ov;
#pragma unroll
      for (int i = 0; i < 8; ++i) ov[i] = (__bf16)o[i];
      *(v8bf*)(qb + qr * QS + cbase + d0) = ov;
    }
  }
  __syncthreads();

  // ---- masked mean over chars (pooling commuted ahead of Wo) ----
  float cnt = 0.f;
#pragma unroll
  for (int r = 0; r < LCH; ++r) cnt += (chars[r] != 0) ? 1.f : 0.f;
  for (int c = tid; c < DMODEL; c += 256) {
    float s = 0.f;
#pragma unroll
    for (int r = 0; r < LCH; ++r)
      if (chars[r] != 0) s += (float)qb[r * QS + c];
    pooled[(size_t)word * DMODEL + c] = s / cnt;
  }
}

// ---------------------------------------------------------------------------
// head: group-mean over 4 words, then name_ft@Wo -> tanh(.@W1) -> tanh(.@W2)
// fused in LDS; 32 names per block, 64 blocks.
// ---------------------------------------------------------------------------
__global__ __launch_bounds__(256) void head_kernel(
    const float*  __restrict__ pooled,   // [8192][512]
    const int*    __restrict__ n_words,  // [2048]
    const __bf16* __restrict__ WoT,      // [512][512]
    const __bf16* __restrict__ W1T,      // [512][512]
    const __bf16* __restrict__ W2T,      // [256][512]
    float*        __restrict__ out)      // [2048][256]
{
  __shared__ alignas(32) __bf16 Abuf[32 * XS];
  __shared__ alignas(32) __bf16 Bbuf[32 * XS];

  const int base = blockIdx.x * 32;
  const int tid  = threadIdx.x;
  const int wave = tid >> 5, lane = tid & 31;

  // group mean of 4 pooled word vectors -> bf16 A tile
  for (int i = tid; i < 32 * DMODEL; i += 256) {
    int r = i >> 9, c = i & 511;
    int g = base + r;
    const float* p = pooled + (size_t)(g * WPN) * DMODEL + c;
    float s = 0.f;
#pragma unroll
    for (int w = 0; w < WPN; ++w) s += p[(size_t)w * DMODEL];
    Abuf[r * XS + c] = (__bf16)(s / (float)n_words[g]);
  }
  __syncthreads();

  gemm_tile_512(Abuf, WoT, Bbuf, XS, wave, lane, false);  // name_ft (linear)
  __syncthreads();
  gemm_tile_512(Bbuf, W1T, Abuf, XS, wave, lane, true);   // h = tanh(.@W1)
  __syncthreads();

  // out = tanh(h @ W2): N=256, each wave owns 32 cols (2 tiles of 16)
  {
    const int lm   = lane & 15;
    const int koff = (lane < 16) ? 0 : 16;
    const int bko  = (lane < 16) ? 0 : 16;
    const int hi8  = (lane >> 4) * 8;
#pragma unroll
    for (int t = 0; t < 2; ++t) {
      const int ncol = wave * 32 + t * 16 + lm;
      const __bf16* wrow = W2T + (size_t)ncol * DMODEL + bko;
      v8f acc0 = {};
      v8f acc1 = {};
      for (int kk = 0; kk < 16; ++kk) {
        const int kb0 = kk * 32;
        v16bf b  = *(const v16bf*)(wrow + kb0);
        v16bf a0 = load_afrag(Abuf + lm * XS + kb0 + koff);
        v16bf a1 = load_afrag(Abuf + (lm + 16) * XS + kb0 + koff);
        acc0 = __builtin_amdgcn_wmma_f32_16x16x32_bf16(false, a0, false, b,
                                                       (short)0, acc0, false, false);
        acc1 = __builtin_amdgcn_wmma_f32_16x16x32_bf16(false, a1, false, b,
                                                       (short)0, acc1, false, false);
      }
#pragma unroll
      for (int r = 0; r < 8; ++r) {
        out[(size_t)(base + r + hi8) * DOUTC + ncol]      = tanhf(acc0[r]);
        out[(size_t)(base + 16 + r + hi8) * DOUTC + ncol] = tanhf(acc1[r]);
      }
    }
  }
}

// ---------------------------------------------------------------------------
extern "C" void kernel_launch(void* const* d_in, const int* in_sizes, int n_in,
                              void* d_out, int out_size, void* d_ws, size_t ws_size,
                              hipStream_t stream)
{
  (void)in_sizes; (void)n_in; (void)out_size; (void)ws_size;
  const int*   inputs  = (const int*)d_in[0];
  const int*   n_words = (const int*)d_in[1];
  /* d_in[2] = n_names: output regrouping is view-only, unused */
  const float* emb = (const float*)d_in[3];
  const float* Wq  = (const float*)d_in[4];
  const float* Wk  = (const float*)d_in[5];
  const float* Wv  = (const float*)d_in[6];
  const float* Wo  = (const float*)d_in[7];
  const float* W1  = (const float*)d_in[8];
  const float* W2  = (const float*)d_in[9];
  float* out = (float*)d_out;

  char* ws = (char*)d_ws;
  size_t off = 0;
  __bf16* embB = (__bf16*)(ws + off); off += (size_t)VOCABC * DMODEL * 2;
  __bf16* WqT  = (__bf16*)(ws + off); off += (size_t)DMODEL * DMODEL * 2;
  __bf16* WkT  = (__bf16*)(ws + off); off += (size_t)DMODEL * DMODEL * 2;
  __bf16* WvT  = (__bf16*)(ws + off); off += (size_t)DMODEL * DMODEL * 2;
  __bf16* WoT  = (__bf16*)(ws + off); off += (size_t)DMODEL * DMODEL * 2;
  __bf16* W1T  = (__bf16*)(ws + off); off += (size_t)DMODEL * DMODEL * 2;
  __bf16* W2T  = (__bf16*)(ws + off); off += (size_t)DOUTC * DMODEL * 2;
  float*  pooled = (float*)(ws + off); off += (size_t)N_WORDSC * DMODEL * 4;

  convert_kernel<<<(VOCABC * DMODEL + 255) / 256, 256, 0, stream>>>(
      emb, embB, VOCABC * DMODEL);
  transpose_convert_kernel<<<(DMODEL * DMODEL + 255) / 256, 256, 0, stream>>>(
      Wq, WqT, DMODEL, DMODEL);
  transpose_convert_kernel<<<(DMODEL * DMODEL + 255) / 256, 256, 0, stream>>>(
      Wk, WkT, DMODEL, DMODEL);
  transpose_convert_kernel<<<(DMODEL * DMODEL + 255) / 256, 256, 0, stream>>>(
      Wv, WvT, DMODEL, DMODEL);
  transpose_convert_kernel<<<(DMODEL * DMODEL + 255) / 256, 256, 0, stream>>>(
      Wo, WoT, DMODEL, DMODEL);
  transpose_convert_kernel<<<(DMODEL * DMODEL + 255) / 256, 256, 0, stream>>>(
      W1, W1T, DMODEL, DMODEL);
  transpose_convert_kernel<<<(DMODEL * DOUTC + 255) / 256, 256, 0, stream>>>(
      W2, W2T, DMODEL, DOUTC);

  encoder_kernel<<<N_WORDSC, 256, 0, stream>>>(inputs, embB, WqT, WkT, WvT, pooled);
  head_kernel<<<N_NAMESC / 32, 256, 0, stream>>>(pooled, n_words, WoT, W1T, W2T, out);
}
